// MultiHeadAttention_9337258902461
// MI455X (gfx1250) — matrix operand order repaired and checked by reading, not compiled, not run on hardware
//
#include <hip/hip_runtime.h>

#define NB 2
#define NT 2048
#define NS 2048
#define ND 1024
#define NH 16
#define NDK 64
#define NHD 1024            // H*DK
#define NR (NB*NT)          // 4096 rows

typedef __bf16 bf16_t;
typedef __attribute__((ext_vector_type(16))) __bf16 v16bf;
typedef __attribute__((ext_vector_type(8)))  __bf16 v8bf;
typedef __attribute__((ext_vector_type(8)))  float  v8f;

// A-fragment (16x32 bf16): lane holds row = lane%16.
// Contiguous K order: lanes 0-15 hold K=0..15 ; lanes 16-31 hold K=16..31.
static __device__ __forceinline__ v16bf load_a_frag(const bf16_t* chunk32, int half) {
    v8bf lo = *(const v8bf*)(chunk32 + half * 16);
    v8bf hi = *(const v8bf*)(chunk32 + half * 16 + 8);
    v16bf r;
#pragma unroll
    for (int i = 0; i < 8; ++i) { r[i] = lo[i]; r[8 + i] = hi[i]; }
    return r;
}

static __device__ __forceinline__ v8f wmma_bf16(v16bf a, v16bf b, v8f c) {
    return __builtin_amdgcn_wmma_f32_16x16x32_bf16(
        /*neg_a=*/false, a, /*neg_b=*/false, b,
        /*c_mod=*/(short)0, c, /*reuse_a=*/false, /*reuse_b=*/false);
}

// Async DMA of 16 bytes/lane from global into LDS (ASYNCcnt-tracked).
static __device__ __forceinline__ void async_ld16(const void* gaddr, void* ldsaddr) {
    unsigned lds_off = (unsigned)(unsigned long long)ldsaddr;  // LDS aperture: low 32 bits = LDS offset
    asm volatile("global_load_async_to_lds_b128 %0, %1, off"
                 :: "v"(lds_off), "v"(gaddr) : "memory");
}

__global__ __launch_bounds__(256) void convert_f32_bf16(
        const float* __restrict__ in, bf16_t* __restrict__ out, int n) {
    int i = blockIdx.x * 256 + threadIdx.x;
    if (i < n) out[i] = (bf16_t)in[i];
}

// out[c*R + r] = bf16(in[r*C + c])  (store weights column-major for B-fragments)
__global__ __launch_bounds__(256) void transpose_f32_bf16(
        const float* __restrict__ in, bf16_t* __restrict__ out, int R, int C) {
    int i = blockIdx.x * 256 + threadIdx.x;
    if (i < R * C) {
        int c = i / R;
        int r = i - c * R;
        out[i] = (bf16_t)in[r * C + c];
    }
}

// Fused QKV projection: Y = X * W + bias ; q additionally scaled by 1/sqrt(DK).
// Register-blocked 32x64 per wave: 2 A-frags x 4 B-frags -> 8 WMMAs per k-step.
// Q,K stored (B,H,seq,DK); V stored transposed (B,H,DK,S).
__global__ __launch_bounds__(256) void qkv_proj(
        const bf16_t* __restrict__ Xq, const bf16_t* __restrict__ Xk, const bf16_t* __restrict__ Xv,
        const bf16_t* __restrict__ WqT, const bf16_t* __restrict__ WkT, const bf16_t* __restrict__ WvT,
        const float* __restrict__ bq, const float* __restrict__ bk, const float* __restrict__ bv,
        bf16_t* __restrict__ Qo, bf16_t* __restrict__ Ko, bf16_t* __restrict__ Vto) {
    const int which = blockIdx.y;                       // 0=q 1=k 2=v
    const bf16_t* X   = which == 0 ? Xq : (which == 1 ? Xk : Xv);
    const bf16_t* WT  = which == 0 ? WqT : (which == 1 ? WkT : WvT);
    const float* bias = which == 0 ? bq : (which == 1 ? bk : bv);

    const int wave = blockIdx.x * 8 + (threadIdx.x >> 5);  // 0..2047
    const int rt = wave >> 4;                              // 128 row blocks of 32
    const int ct = wave & 15;                              // 16 col blocks of 64
    const int lane = threadIdx.x & 31;
    const int nlo = lane & 15, half = lane >> 4;
    const int row0 = rt * 32, col0 = ct * 64;

    const bf16_t* xr[2];
    const bf16_t* wr[4];
#pragma unroll
    for (int r = 0; r < 2; ++r) xr[r] = X  + (size_t)(row0 + r * 16 + nlo) * ND;
#pragma unroll
    for (int c = 0; c < 4; ++c) wr[c] = WT + (size_t)(col0 + c * 16 + nlo) * ND;

    v8f acc[2][4];
#pragma unroll
    for (int r = 0; r < 2; ++r)
#pragma unroll
        for (int c = 0; c < 4; ++c) acc[r][c] = (v8f){};

#pragma unroll 2
    for (int k = 0; k < ND; k += 32) {
        v16bf a0 = load_a_frag(xr[0] + k, half);
        v16bf a1 = load_a_frag(xr[1] + k, half);
        v16bf b0 = *(const v16bf*)(wr[0] + k + half * 16);
        v16bf b1 = *(const v16bf*)(wr[1] + k + half * 16);
        v16bf b2 = *(const v16bf*)(wr[2] + k + half * 16);
        v16bf b3 = *(const v16bf*)(wr[3] + k + half * 16);
        acc[0][0] = wmma_bf16(a0, b0, acc[0][0]);
        acc[0][1] = wmma_bf16(a0, b1, acc[0][1]);
        acc[0][2] = wmma_bf16(a0, b2, acc[0][2]);
        acc[0][3] = wmma_bf16(a0, b3, acc[0][3]);
        acc[1][0] = wmma_bf16(a1, b0, acc[1][0]);
        acc[1][1] = wmma_bf16(a1, b1, acc[1][1]);
        acc[1][2] = wmma_bf16(a1, b2, acc[1][2]);
        acc[1][3] = wmma_bf16(a1, b3, acc[1][3]);
    }

    const float scale = (which == 0) ? 0.125f : 1.0f;   // 1/sqrt(64)
    const int h = col0 >> 6;                            // col0 is 64-aligned
#pragma unroll
    for (int c = 0; c < 4; ++c) {
        const int dk = c * 16 + nlo;
        const float bs = bias[col0 + dk];
#pragma unroll
        for (int r = 0; r < 2; ++r) {
#pragma unroll
            for (int i = 0; i < 8; ++i) {
                int rr = row0 + r * 16 + i + 8 * half;
                int b_ = rr >> 11, t = rr & (NT - 1);
                float v = (acc[r][c][i] + bs) * scale;
                size_t bhh = (size_t)(b_ * NH + h);
                if (which == 2)      Vto[(bhh * NDK + dk) * NS + t] = (bf16_t)v;
                else if (which == 1) Ko[(bhh * NT + t) * NDK + dk]  = (bf16_t)v;
                else                 Qo[(bhh * NT + t) * NDK + dk]  = (bf16_t)v;
            }
        }
    }
}

// Flash attention: block = 8 waves x 16 query rows of one (b,h).
// K/V chunks staged into LDS via async-to-LDS DMA, double buffered, shared by all waves.
__global__ __launch_bounds__(256) void flash_attn(
        const bf16_t* __restrict__ Q, const bf16_t* __restrict__ K,
        const bf16_t* __restrict__ Vt, bf16_t* __restrict__ O) {
    __shared__ __align__(64) __bf16 Ktile[2][32][64];   // 2 x 4KB
    __shared__ __align__(64) __bf16 Vtile[2][64][32];   // 2 x 4KB (dk-major)
    __shared__ __align__(64) __bf16 Ptile[8][16][32];   // 8KB, per-wave

    const int tid = threadIdx.x;
    const int lane = tid & 31;
    const int w = tid >> 5;
    const int nlo = lane & 15, half = lane >> 4;
    const int bh = blockIdx.y;                          // b*H + h
    const int t0 = (blockIdx.x * 8 + w) * 16;

    const bf16_t* Kb = K  + (size_t)bh * NS * NDK;
    const bf16_t* Vb = Vt + (size_t)bh * NDK * NS;

    // cooperative DMA assignment: each of 256 threads moves 16B of K + 16B of V per chunk
    const int kr = tid >> 3, kc = (tid & 7) * 8;        // K: 32 rows x 64 bf16
    const int vr = tid >> 2, vc = (tid & 3) * 8;        // V: 64 rows x 32 bf16

    const bf16_t* qrow = Q + ((size_t)bh * NT + t0 + nlo) * NDK;
    v16bf q0 = load_a_frag(qrow, half);                 // dk 0..31
    v16bf q1 = load_a_frag(qrow + 32, half);            // dk 32..63

    float m_run[8], d_run[8];
    v8f o0 = {}, o1 = {}, o2 = {}, o3 = {};
#pragma unroll
    for (int i = 0; i < 8; ++i) { m_run[i] = -3.0e38f; d_run[i] = 0.f; }

    // prologue: stage chunk 0 into buffer 0
    async_ld16(Kb + (size_t)kr * NDK + kc, &Ktile[0][kr][kc]);
    async_ld16(Vb + (size_t)vr * NS + vc,  &Vtile[0][vr][vc]);

    for (int s0 = 0; s0 < NS; s0 += 32) {
        const int buf = (s0 >> 5) & 1;
        if (s0 + 32 < NS) {
            // prefetch next chunk into the other buffer, then wait for current (in-order)
            async_ld16(Kb + (size_t)(s0 + 32 + kr) * NDK + kc, &Ktile[buf ^ 1][kr][kc]);
            async_ld16(Vb + (size_t)vr * NS + (s0 + 32) + vc,  &Vtile[buf ^ 1][vr][vc]);
            asm volatile("s_wait_asynccnt 0x2" ::: "memory");
        } else {
            asm volatile("s_wait_asynccnt 0x0" ::: "memory");
        }
        __syncthreads();

        // scores: C(M=t, N=s), two 16-col s-subtiles, K-dim = 64
        const bf16_t* krow0 = &Ktile[buf][nlo][half * 16];
        const bf16_t* krow1 = &Ktile[buf][16 + nlo][half * 16];
        v8f c0 = {}, c1 = {};
        c0 = wmma_bf16(q0, *(const v16bf*)(krow0),      c0);
        c0 = wmma_bf16(q1, *(const v16bf*)(krow0 + 32), c0);
        c1 = wmma_bf16(q0, *(const v16bf*)(krow1),      c1);
        c1 = wmma_bf16(q1, *(const v16bf*)(krow1 + 32), c1);

        // online softmax per row (row i+8*half lives across the 16-lane half)
#pragma unroll
        for (int i = 0; i < 8; ++i) {
            float rm = fmaxf(c0[i], c1[i]);
            rm = fmaxf(rm, __shfl_xor(rm, 1, 32));
            rm = fmaxf(rm, __shfl_xor(rm, 2, 32));
            rm = fmaxf(rm, __shfl_xor(rm, 4, 32));
            rm = fmaxf(rm, __shfl_xor(rm, 8, 32));
            float mnew = fmaxf(m_run[i], rm);
            float corr = __expf(m_run[i] - mnew);
            float p0 = __expf(c0[i] - mnew);
            float p1 = __expf(c1[i] - mnew);
            float rs = p0 + p1;
            rs += __shfl_xor(rs, 1, 32);
            rs += __shfl_xor(rs, 2, 32);
            rs += __shfl_xor(rs, 4, 32);
            rs += __shfl_xor(rs, 8, 32);
            d_run[i] = d_run[i] * corr + rs;
            m_run[i] = mnew;
            o0[i] *= corr; o1[i] *= corr; o2[i] *= corr; o3[i] *= corr;
            Ptile[w][i + 8 * half][nlo]      = (bf16_t)p0;
            Ptile[w][i + 8 * half][nlo + 16] = (bf16_t)p1;
        }
        asm volatile("s_wait_dscnt 0x0" ::: "memory");

        // O += P(16x32) x V(32x64): A from per-wave LDS P, B from staged V tile
        v16bf ap = load_a_frag(&Ptile[w][nlo][0], half);
        o0 = wmma_bf16(ap, *(const v16bf*)&Vtile[buf][nlo][half * 16],      o0);
        o1 = wmma_bf16(ap, *(const v16bf*)&Vtile[buf][16 + nlo][half * 16], o1);
        o2 = wmma_bf16(ap, *(const v16bf*)&Vtile[buf][32 + nlo][half * 16], o2);
        o3 = wmma_bf16(ap, *(const v16bf*)&Vtile[buf][48 + nlo][half * 16], o3);

        __syncthreads();   // protect buffer `buf` before next prefetch overwrites it
    }

    const int b_ = bh >> 4, h = bh & 15;
#pragma unroll
    for (int i = 0; i < 8; ++i) {
        float inv = 1.0f / d_run[i];
        int t = t0 + i + 8 * half;
        bf16_t* orow = O + (size_t)(b_ * NT + t) * NHD + h * NDK;
        orow[nlo]      = (bf16_t)(o0[i] * inv);
        orow[16 + nlo] = (bf16_t)(o1[i] * inv);
        orow[32 + nlo] = (bf16_t)(o2[i] * inv);
        orow[48 + nlo] = (bf16_t)(o3[i] * inv);
    }
}

// out = O(4096x1024) x Wo(1024x1024) + bo, fp32 output; 32x64 register blocking.
__global__ __launch_bounds__(256) void out_proj(
        const bf16_t* __restrict__ Ob, const bf16_t* __restrict__ WoT,
        const float* __restrict__ bo, float* __restrict__ out) {
    const int wave = blockIdx.x * 8 + (threadIdx.x >> 5);  // 0..2047
    const int rt = wave >> 4, ct = wave & 15;
    const int lane = threadIdx.x & 31;
    const int nlo = lane & 15, half = lane >> 4;
    const int row0 = rt * 32, col0 = ct * 64;

    const bf16_t* xr[2];
    const bf16_t* wr[4];
#pragma unroll
    for (int r = 0; r < 2; ++r) xr[r] = Ob  + (size_t)(row0 + r * 16 + nlo) * NHD;
#pragma unroll
    for (int c = 0; c < 4; ++c) wr[c] = WoT + (size_t)(col0 + c * 16 + nlo) * NHD;

    v8f acc[2][4];
#pragma unroll
    for (int r = 0; r < 2; ++r)
#pragma unroll
        for (int c = 0; c < 4; ++c) acc[r][c] = (v8f){};

#pragma unroll 2
    for (int k = 0; k < NHD; k += 32) {
        v16bf a0 = load_a_frag(xr[0] + k, half);
        v16bf a1 = load_a_frag(xr[1] + k, half);
        v16bf b0 = *(const v16bf*)(wr[0] + k + half * 16);
        v16bf b1 = *(const v16bf*)(wr[1] + k + half * 16);
        v16bf b2 = *(const v16bf*)(wr[2] + k + half * 16);
        v16bf b3 = *(const v16bf*)(wr[3] + k + half * 16);
        acc[0][0] = wmma_bf16(a0, b0, acc[0][0]);
        acc[0][1] = wmma_bf16(a0, b1, acc[0][1]);
        acc[0][2] = wmma_bf16(a0, b2, acc[0][2]);
        acc[0][3] = wmma_bf16(a0, b3, acc[0][3]);
        acc[1][0] = wmma_bf16(a1, b0, acc[1][0]);
        acc[1][1] = wmma_bf16(a1, b1, acc[1][1]);
        acc[1][2] = wmma_bf16(a1, b2, acc[1][2]);
        acc[1][3] = wmma_bf16(a1, b3, acc[1][3]);
    }

#pragma unroll
    for (int c = 0; c < 4; ++c) {
        const int col = col0 + c * 16 + nlo;
        const float bs = bo[col];
#pragma unroll
        for (int r = 0; r < 2; ++r) {
#pragma unroll
            for (int i = 0; i < 8; ++i) {
                int rr = row0 + r * 16 + i + 8 * half;
                out[(size_t)rr * ND + col] = acc[r][c][i] + bs;
            }
        }
    }
}

extern "C" void kernel_launch(void* const* d_in, const int* in_sizes, int n_in,
                              void* d_out, int out_size, void* d_ws, size_t ws_size,
                              hipStream_t stream) {
    const float* q_in = (const float*)d_in[0];
    const float* v_in = (const float*)d_in[1];
    const float* k_in = (const float*)d_in[2];
    const float* Wq   = (const float*)d_in[3];
    const float* bq   = (const float*)d_in[4];
    const float* Wk   = (const float*)d_in[5];
    const float* bk   = (const float*)d_in[6];
    const float* Wv   = (const float*)d_in[7];
    const float* bv   = (const float*)d_in[8];
    const float* Wo   = (const float*)d_in[9];
    const float* bo   = (const float*)d_in[10];
    float* out = (float*)d_out;

    char* ws = (char*)d_ws;
    const size_t EIN = (size_t)NR * ND;   // 4,194,304 elements
    const size_t EW  = (size_t)ND * NHD;  // 1,048,576 elements
    bf16_t* qin = (bf16_t*)(ws);
    bf16_t* kin = (bf16_t*)(ws + EIN * 2);
    bf16_t* vin = (bf16_t*)(ws + EIN * 4);
    bf16_t* wqt = (bf16_t*)(ws + EIN * 6);
    bf16_t* wkt = (bf16_t*)(ws + EIN * 6 + EW * 2);
    bf16_t* wvt = (bf16_t*)(ws + EIN * 6 + EW * 4);
    bf16_t* wot = (bf16_t*)(ws + EIN * 6 + EW * 6);
    bf16_t* Qb  = (bf16_t*)(ws + EIN * 6 + EW * 8);
    bf16_t* Kb  = (bf16_t*)(ws + EIN * 8 + EW * 8);
    bf16_t* Vtb = (bf16_t*)(ws + EIN * 10 + EW * 8);
    bf16_t* Ob  = (bf16_t*)(ws + EIN * 12 + EW * 8);   // ends at 64 MB

    const int gcv = (int)((EIN + 255) / 256);
    const int gtw = (int)((EW + 255) / 256);
    convert_f32_bf16<<<gcv, 256, 0, stream>>>(q_in, qin, (int)EIN);
    convert_f32_bf16<<<gcv, 256, 0, stream>>>(k_in, kin, (int)EIN);
    convert_f32_bf16<<<gcv, 256, 0, stream>>>(v_in, vin, (int)EIN);
    transpose_f32_bf16<<<gtw, 256, 0, stream>>>(Wq, wqt, ND, NHD);
    transpose_f32_bf16<<<gtw, 256, 0, stream>>>(Wk, wkt, ND, NHD);
    transpose_f32_bf16<<<gtw, 256, 0, stream>>>(Wv, wvt, ND, NHD);
    transpose_f32_bf16<<<gtw, 256, 0, stream>>>(Wo, wot, NHD, ND);

    qkv_proj<<<dim3(256, 3), 256, 0, stream>>>(qin, kin, vin, wqt, wkt, wvt,
                                               bq, bk, bv, Qb, Kb, Vtb);
    flash_attn<<<dim3(16, 32), 256, 0, stream>>>(Qb, Kb, Vtb, Ob);
    out_proj<<<256, 256, 0, stream>>>(Ob, wot, bo, out);
}
